// OmicsEmbeddingLayer_83296595738829
// MI455X (gfx1250) — compile-verified, matrix operand order checked by reading, not run on hardware
//
#include <hip/hip_runtime.h>
#include <stdint.h>

typedef _Float16 v16h __attribute__((ext_vector_type(16)));
typedef float    v8f  __attribute__((ext_vector_type(8)));
typedef _Float16 h4   __attribute__((ext_vector_type(4)));

#define B_DIM 16384
#define L_DIM 4096
#define G_DIM 30000
#define H_DIM 256

#define M_TILE 64
#define KB     32
#define STRA   40    // sA row stride (f16 elems), padded vs 32 to dodge bank conflicts
#define STRB   40    // sB row stride (f16 elems)
#define HSTR   264   // sH row stride (f16 elems)
#define OSTR   260   // sOut row stride (f32 elems)

// LDS carve (bytes)
#define SA_OFF   0          // 2 x (64*40*2  = 5120)
#define SA_SZ    5120
#define SB_OFF   10240      // 2 x (256*40*2 = 20480)
#define SB_SZ    20480
#define SH_OFF   51200      // 64*264*2 = 33792
#define SMEM_BYTES 84992
// phase-3 overlay: sOut f32 64*260*4 = 66560 @0, stats @66560

// ---------------- prep: gather gene embeddings (transposed, f16) + W1^T f16 ----------------
__global__ __launch_bounds__(256) void prep_kernel(
    const float* __restrict__ emb, const float* __restrict__ W1,
    const int* __restrict__ gidx, _Float16* __restrict__ featT,
    _Float16* __restrict__ W1T)
{
  const int b = blockIdx.x;
  const int t = threadIdx.x;
  if (b < L_DIM) {
    const int k = b;
    const int g = gidx[k];
    // featT[n][k] = emb[gidx[k]][n]
    featT[(size_t)t * L_DIM + k] = (_Float16)emb[(size_t)g * H_DIM + t];
  } else {
    const int n = b - L_DIM;
    // W1T[n][k] = W1[k][n]
    W1T[(size_t)n * H_DIM + t] = (_Float16)W1[(size_t)t * H_DIM + n];
  }
}

// async DMA: global (64B row) -> LDS, tracked by ASYNCcnt; INST_OFFSET applies to
// both LDS and global addresses (ISA 10. async pseudocode), so one addr pair covers 4x16B.
__device__ __forceinline__ void async_row64(unsigned lds_off, unsigned long long gaddr) {
  asm volatile("global_load_async_to_lds_b128 %0, %1, off"
               :: "v"(lds_off), "v"(gaddr) : "memory");
  asm volatile("global_load_async_to_lds_b128 %0, %1, off offset:16"
               :: "v"(lds_off), "v"(gaddr) : "memory");
  asm volatile("global_load_async_to_lds_b128 %0, %1, off offset:32"
               :: "v"(lds_off), "v"(gaddr) : "memory");
  asm volatile("global_load_async_to_lds_b128 %0, %1, off offset:48"
               :: "v"(lds_off), "v"(gaddr) : "memory");
}
__device__ __forceinline__ void wait_async0() {
  asm volatile("s_wait_asynccnt 0x0" ::: "memory");
}

// ---------------- fused: x@feat -> @W1 + b1 -> ReLU -> LayerNorm ----------------
__global__ __launch_bounds__(256) void fused_kernel(
    const float* __restrict__ x, const float* __restrict__ b1,
    const float* __restrict__ gamma, const float* __restrict__ beta,
    const _Float16* __restrict__ featT, const _Float16* __restrict__ W1T,
    float* __restrict__ out)
{
  __shared__ char smem[SMEM_BYTES];
  _Float16* sH = (_Float16*)(smem + SH_OFF);
  // LDS byte offset of smem within the wave allocation (generic ptr low 32 bits)
  const unsigned sbase = (unsigned)(uintptr_t)(&smem[0]);

  const int t      = threadIdx.x;
  const int lane   = t & 31;
  const int wid    = t >> 5;
  const int wave_m = wid & 1;   // 0..1 -> 32-row slab
  const int wave_n = wid >> 1;  // 0..3 -> 64-col slab
  const int m0     = blockIdx.x * M_TILE;

  // fragment addressing: lanes 0-15 hold K runs {0..7,16..23}, lanes 16-31 {8..15,24..31}
  const int frow = lane & 15;
  const int fko  = (lane < 16) ? 0 : 8;

  // staging split: A tile 64x32 f32->f16 (4 threads/row, 8 floats each)
  const int arow = t >> 2;
  const int acg  = (t & 3) * 8;

  auto stageA = [&](_Float16* dst, int k0) {
    const float* p = x + (size_t)(m0 + arow) * L_DIM + k0 + acg;
    float4 f0 = *(const float4*)p;
    float4 f1 = *(const float4*)(p + 4);
    if (k0 + 4 * KB < L_DIM) __builtin_prefetch(p + 4 * KB, 0, 0);  // global_prefetch_b8
    h4 h0 = { (_Float16)f0.x, (_Float16)f0.y, (_Float16)f0.z, (_Float16)f0.w };
    h4 h1 = { (_Float16)f1.x, (_Float16)f1.y, (_Float16)f1.z, (_Float16)f1.w };
    *(h4*)(dst + arow * STRA + acg)     = h0;
    *(h4*)(dst + arow * STRA + acg + 4) = h1;
  };
  // async B staging: sB[n][0..31] = src[n][k0..k0+31], n = t (64 bytes, DMA to LDS)
  auto stageB = [&](unsigned dstOff, const _Float16* src, int srcstride, int k0) {
    const unsigned long long g =
        (unsigned long long)(uintptr_t)(src + (size_t)t * srcstride + k0);
    const unsigned l = sbase + dstOff + (unsigned)(t * (STRB * 2));
    async_row64(l, g);
  };

  auto loadFrag = [&](const _Float16* base, int stride, int mrow, int kbase) -> v16h {
    union { v16h h; uint4 q[2]; } u;
    const _Float16* p = base + (size_t)(mrow + frow) * stride + kbase + fko;
    u.q[0] = *(const uint4*)p;
    u.q[1] = *(const uint4*)(p + 16);
    return u.h;
  };

  const v8f vzero = {0.f,0.f,0.f,0.f,0.f,0.f,0.f,0.f};
  v8f acc[2][4] = { {vzero,vzero,vzero,vzero}, {vzero,vzero,vzero,vzero} };

  // ---------------- GEMM1: h = x @ feat, K = 4096, double-buffered ----------------
  const int NT = L_DIM / KB;  // 128
  stageA((_Float16*)(smem + SA_OFF), 0);
  stageB(SB_OFF, featT, L_DIM, 0);
  wait_async0();
  __syncthreads();

  for (int kt = 0; kt < NT; ++kt) {
    const int cur = kt & 1;
    _Float16* curA = (_Float16*)(smem + SA_OFF + cur * SA_SZ);
    _Float16* curB = (_Float16*)(smem + SB_OFF + cur * SB_SZ);
    if (kt + 1 < NT) {
      _Float16* nxtA = (_Float16*)(smem + SA_OFF + (cur ^ 1) * SA_SZ);
      stageA(nxtA, (kt + 1) * KB);
      stageB(SB_OFF + (cur ^ 1) * SB_SZ, featT, L_DIM, (kt + 1) * KB);
    }

    v16h a0 = loadFrag(curA, STRA, wave_m * 32 + 0,  0);
    v16h a1 = loadFrag(curA, STRA, wave_m * 32 + 16, 0);
    v16h bf[4];
    #pragma unroll
    for (int nj = 0; nj < 4; ++nj)
      bf[nj] = loadFrag(curB, STRB, wave_n * 64 + nj * 16, 0);
    #pragma unroll
    for (int nj = 0; nj < 4; ++nj) {
      acc[0][nj] = __builtin_amdgcn_wmma_f32_16x16x32_f16(false, a0, false, bf[nj], (short)0, acc[0][nj], false, false);
      acc[1][nj] = __builtin_amdgcn_wmma_f32_16x16x32_f16(false, a1, false, bf[nj], (short)0, acc[1][nj], false, false);
    }
    wait_async0();   // next-buffer DMA complete before the barrier publishes it
    __syncthreads();
  }

  // ---------------- spill h tile to LDS as f16 (A operand of GEMM2) ----------------
  #pragma unroll
  for (int mi = 0; mi < 2; ++mi) {
    #pragma unroll
    for (int nj = 0; nj < 4; ++nj) {
      const int rbase = wave_m * 32 + mi * 16 + ((lane < 16) ? 0 : 8);
      const int col   = wave_n * 64 + nj * 16 + (lane & 15);
      #pragma unroll
      for (int r = 0; r < 8; ++r)
        sH[(size_t)(rbase + r) * HSTR + col] = (_Float16)acc[mi][nj][r];
    }
  }
  __syncthreads();

  // ---------------- GEMM2: h2 = h @ W1, K = 256 ----------------
  _Float16* sB0 = (_Float16*)(smem + SB_OFF);
  v8f acc2[2][4] = { {vzero,vzero,vzero,vzero}, {vzero,vzero,vzero,vzero} };
  for (int kt2 = 0; kt2 < H_DIM / KB; ++kt2) {
    const int k0 = kt2 * KB;
    stageB(SB_OFF, W1T, H_DIM, k0);
    wait_async0();
    __syncthreads();
    v16h a0 = loadFrag(sH, HSTR, wave_m * 32 + 0,  k0);
    v16h a1 = loadFrag(sH, HSTR, wave_m * 32 + 16, k0);
    v16h bf[4];
    #pragma unroll
    for (int nj = 0; nj < 4; ++nj)
      bf[nj] = loadFrag(sB0, STRB, wave_n * 64 + nj * 16, 0);
    #pragma unroll
    for (int nj = 0; nj < 4; ++nj) {
      acc2[0][nj] = __builtin_amdgcn_wmma_f32_16x16x32_f16(false, a0, false, bf[nj], (short)0, acc2[0][nj], false, false);
      acc2[1][nj] = __builtin_amdgcn_wmma_f32_16x16x32_f16(false, a1, false, bf[nj], (short)0, acc2[1][nj], false, false);
    }
    __syncthreads();
  }

  // ---------------- bias + ReLU -> f32 tile in LDS (overlay) ----------------
  float* sOut = (float*)smem;                   // 64 x OSTR f32
  float* sMu  = (float*)(smem + 66560);
  float* sRs  = (float*)(smem + 66560 + 256);
  #pragma unroll
  for (int mi = 0; mi < 2; ++mi) {
    #pragma unroll
    for (int nj = 0; nj < 4; ++nj) {
      const int col   = wave_n * 64 + nj * 16 + (lane & 15);
      const float bias = b1[col];
      const int rbase = wave_m * 32 + mi * 16 + ((lane < 16) ? 0 : 8);
      #pragma unroll
      for (int r = 0; r < 8; ++r) {
        float v = acc2[mi][nj][r] + bias;
        v = v > 0.f ? v : 0.f;
        sOut[(size_t)(rbase + r) * OSTR + col] = v;
      }
    }
  }
  __syncthreads();

  // ---------------- LayerNorm stats (one thread per row) ----------------
  if (t < 64) {
    float s = 0.f, ss = 0.f;
    for (int c = 0; c < H_DIM; ++c) {
      float v = sOut[(size_t)t * OSTR + c];
      s += v; ss += v * v;
    }
    const float mu  = s * (1.f / 256.f);
    const float var = ss * (1.f / 256.f) - mu * mu;
    sMu[t] = mu;
    sRs[t] = rsqrtf(var + 1e-5f);
  }
  __syncthreads();

  // ---------------- scale/shift + store ----------------
  const int orow = t >> 2;
  const int oc   = (t & 3) * 64;
  const float mu = sMu[orow];
  const float rs = sRs[orow];
  float* op = out + (size_t)(m0 + orow) * H_DIM + oc;
  #pragma unroll
  for (int c = 0; c < 64; c += 4) {
    float4 v  = *(float4*)(sOut + (size_t)orow * OSTR + oc + c);
    float4 g  = *(const float4*)(gamma + oc + c);
    float4 be = *(const float4*)(beta + oc + c);
    float4 o;
    o.x = g.x * (v.x - mu) * rs + be.x;
    o.y = g.y * (v.y - mu) * rs + be.y;
    o.z = g.z * (v.z - mu) * rs + be.z;
    o.w = g.w * (v.w - mu) * rs + be.w;
    *(float4*)(op + c) = o;
  }
}

extern "C" void kernel_launch(void* const* d_in, const int* in_sizes, int n_in,
                              void* d_out, int out_size, void* d_ws, size_t ws_size,
                              hipStream_t stream) {
  const float* x     = (const float*)d_in[0];
  const float* emb   = (const float*)d_in[1];
  const float* W1    = (const float*)d_in[2];
  const float* b1    = (const float*)d_in[3];
  const float* gamma = (const float*)d_in[4];
  const float* beta  = (const float*)d_in[5];
  const int*   gidx  = (const int*)d_in[6];

  _Float16* featT = (_Float16*)d_ws;                                      // [H][L] f16, 2 MB
  _Float16* W1T   = (_Float16*)((char*)d_ws + (size_t)H_DIM * L_DIM * 2); // [H][H] f16, 128 KB

  prep_kernel<<<L_DIM + H_DIM, 256, 0, stream>>>(emb, W1, gidx, featT, W1T);
  fused_kernel<<<B_DIM / M_TILE, 256, 0, stream>>>(x, b1, gamma, beta, featT, W1T,
                                                   (float*)d_out);
}
